// Pool_45603962749696
// MI455X (gfx1250) — matrix operand order repaired and checked by reading, not compiled, not run on hardware
//
#include <hip/hip_runtime.h>
#include <hip/hip_bf16.h>

typedef __attribute__((ext_vector_type(16))) _Float16 v16h;
typedef __attribute__((ext_vector_type(8)))  float    v8f;

#define NPTS 2048
#define NC   1024
#define NDIM 256
#define NB   4

// ---------------------------------------------------------------------------
// Kernel 1: scores[b, n] = dot(h[b, n, :], W) + bias, via v_wmma_f32_16x16x32_f16.
// One wave (32 threads) per 16-row tile of h. K=256 -> 8 fully-unrolled WMMA
// steps so loads batch into clauses and 8 static v_wmma are emitted.
// A layout (16-bit A 16x32): lane L holds row M=L%16; lanes 0-15: K = kb..kb+15
// ascending; lanes 16-31: K = kb+16..kb+31 (matches B).
// B layout (32x16, row striped across lanes): lane L holds col N=L%16; lanes 0-15
// hold K = kb..kb+15 ascending; lanes 16-31 hold K = kb+16..kb+31.
// W occupies column N==0 only; other columns zero.
// D (16x16 f32): VGPR v, lanes 0-15 -> M=v, N=lane; lanes 16-31 -> M=8+v.
// ---------------------------------------------------------------------------
__global__ __launch_bounds__(32) void scores_wmma_kernel(
    const float* __restrict__ h, const float* __restrict__ W,
    const float* __restrict__ bias, float* __restrict__ scores)
{
    const int tile  = blockIdx.x;          // 0..511  (4 batches * 128 tiles)
    const int batch = tile >> 7;
    const int r0    = (tile & 127) << 4;
    const int lane  = threadIdx.x;
    const int m     = lane & 15;
    const bool hi   = lane >= 16;

    const float* hrow = h + ((size_t)batch * NPTS + (r0 + m)) * NDIM;

    v8f acc = {};
#pragma unroll
    for (int kb = 0; kb < NDIM; kb += 32) {
        const int ksA = kb + (hi ? 16 : 0);
        const int ksB = kb + (hi ? 16 : 0);
        v16h a, bm;
#pragma unroll
        for (int t = 0; t < 8; ++t) {
            a[t]     = (_Float16)hrow[ksA + t];
            a[8 + t] = (_Float16)hrow[ksA + 8 + t];
        }
#pragma unroll
        for (int t = 0; t < 16; ++t) {
            float wv = W[ksB + t];
            bm[t] = (m == 0) ? (_Float16)wv : (_Float16)0.f;
        }
        acc = __builtin_amdgcn_wmma_f32_16x16x32_f16(
            /*neg_a=*/false, a, /*neg_b=*/false, bm,
            /*c_mod=*/(short)0, acc, /*reuse_a=*/false, /*reuse_b=*/false);
    }

    if (m == 0) {
        const float bb = bias[0];
#pragma unroll
        for (int v = 0; v < 8; ++v)
            scores[batch * NPTS + r0 + (hi ? 8 : 0) + v] = acc[v] + bb;
    }
}

// ---------------------------------------------------------------------------
// Kernel 2: full 1-D Lloyd KMeans + center selection, one batch per block.
// 1024 threads, all state in LDS (52 KB of the 320 KB WGP pool).
// Bitonic sort on (value, original-index) pairs reproduces jnp.sort (quantile
// init centers0[i] = s_sorted[2i+1]) and the stable jnp.argsort lower-median.
// Brute-force argmin with strict '<' matches jnp.argmin first-index ties.
// Per-cluster first-closest-member via u64 atomicMin on (dist_bits<<32)|n
// (dist >= 0 so the IEEE bit pattern is order-preserving).
// ---------------------------------------------------------------------------
__global__ __launch_bounds__(1024) void cluster_kernel(
    const float* __restrict__ scores_g, int* __restrict__ ci_g)
{
    __shared__ float sorig[NPTS];
    __shared__ float sval[NPTS];   // sort values; later: [0..NC) = w_sum, [NC..2NC) = ws_sum
    __shared__ int   sidx[NPTS];
    __shared__ float centers[NC];
    __shared__ float cnt[NC];
    __shared__ float sm[NC];       // Lloyd sums; later wmean
    __shared__ int   asg[NPTS];
    __shared__ unsigned long long best[NC];
    __shared__ int   median_idx_s;

    const int b   = blockIdx.x;
    const int tid = threadIdx.x;
    const float* sc = scores_g + b * NPTS;

    for (int n = tid; n < NPTS; n += 1024) {
        float v = sc[n];
        sorig[n] = v; sval[n] = v; sidx[n] = n;
    }
    __syncthreads();

    // Bitonic sort (ascending by (val, idx))
    for (int k = 2; k <= NPTS; k <<= 1) {
        for (int j = k >> 1; j > 0; j >>= 1) {
            for (int t = tid; t < NPTS; t += 1024) {
                int p = t ^ j;
                if (p > t) {
                    bool up = ((t & k) == 0);
                    float av = sval[t], bv = sval[p];
                    int   ai = sidx[t], bi = sidx[p];
                    bool agtb = (av > bv) || (av == bv && ai > bi);
                    if (agtb == up) {
                        sval[t] = bv; sval[p] = av;
                        sidx[t] = bi; sidx[p] = ai;
                    }
                }
            }
            __syncthreads();
        }
    }

    if (tid == 0) median_idx_s = sidx[(NPTS - 1) >> 1];
    centers[tid] = sval[2 * tid + 1];      // quantile init: floor((i+0.5)*N/C) = 2i+1
    __syncthreads();

    // 10 Lloyd iterations
    for (int it = 0; it < 10; ++it) {
        cnt[tid] = 0.f; sm[tid] = 0.f;
        __syncthreads();
        {
            float s0 = sorig[tid], s1 = sorig[tid + 1024];
            int   b0 = 0, b1 = 0;
            float d0 = fabsf(s0 - centers[0]), d1 = fabsf(s1 - centers[0]);
            for (int j = 1; j < NC; ++j) {
                float c = centers[j];
                float e0 = fabsf(s0 - c); if (e0 < d0) { d0 = e0; b0 = j; }
                float e1 = fabsf(s1 - c); if (e1 < d1) { d1 = e1; b1 = j; }
            }
            atomicAdd(&cnt[b0], 1.f); atomicAdd(&sm[b0], s0);
            atomicAdd(&cnt[b1], 1.f); atomicAdd(&sm[b1], s1);
        }
        __syncthreads();
        {
            float c = cnt[tid];
            if (c > 0.f) centers[tid] = sm[tid] / fmaxf(c, 1.f);
        }
        __syncthreads();
    }

    // Final assignment + sigmoid-weighted sums (reuse sval as w_sum / ws_sum)
    cnt[tid] = 0.f; sval[tid] = 0.f; sval[NC + tid] = 0.f;
    best[tid] = 0xFFFFFFFFFFFFFFFFULL;
    __syncthreads();
    for (int n = tid; n < NPTS; n += 1024) {
        float s = sorig[n];
        int bj = 0; float bd = fabsf(s - centers[0]);
        for (int j = 1; j < NC; ++j) {
            float e = fabsf(s - centers[j]);
            if (e < bd) { bd = e; bj = j; }
        }
        asg[n] = bj;
        float w = 1.f / (1.f + expf(-s));
        atomicAdd(&cnt[bj], 1.f);
        atomicAdd(&sval[bj], w);
        atomicAdd(&sval[NC + bj], w * s);
    }
    __syncthreads();
    {
        float ws = sval[tid];
        sm[tid] = sval[NC + tid] / (ws > 0.f ? ws : 1.f);   // wmean
    }
    __syncthreads();
    for (int n = tid; n < NPTS; n += 1024) {
        int j = asg[n];
        float d = fabsf(sorig[n] - sm[j]);
        unsigned long long key =
            ((unsigned long long)__float_as_uint(d) << 32) | (unsigned int)n;
        atomicMin(&best[j], key);
    }
    __syncthreads();
    {
        int idx = (cnt[tid] > 0.f) ? (int)(best[tid] & 0xFFFFFFFFu) : median_idx_s;
        ci_g[b * NC + tid] = idx;
    }
}

// ---------------------------------------------------------------------------
// Kernel 3: gather outputs (the roofline-critical ~60 MB of traffic).
// out[0 .. B*C*C)       = new_g[b,i,j] = g[b, ci[i], ci[j]]
// out[B*C*C .. +B*C*D)  = new_h[b,i,d] = h[b, ci[i], d]
// All flat indices fit in int32 (g: 16.7M elems, out: 5.2M) -> pure 32-bit
// index math; fastest output dim maps to threadIdx.x for coalesced stores.
// ---------------------------------------------------------------------------
__global__ __launch_bounds__(256) void gather_kernel(
    const float* __restrict__ g, const float* __restrict__ h,
    const int* __restrict__ ci, float* __restrict__ out)
{
    const int GT = NB * NC * NC;                 // 4,194,304
    int e = blockIdx.x * 256 + threadIdx.x;      // < 5,242,880
    if (e < GT) {
        int b = e >> 20;
        int i = (e >> 10) & (NC - 1);
        int j = e & (NC - 1);
        int ri = ci[b * NC + i];
        int rj = ci[b * NC + j];
        out[e] = g[(b * NPTS + ri) * NPTS + rj];
    } else {
        int e2 = e - GT;
        int b = e2 >> 18;
        int i = (e2 >> 8) & (NC - 1);
        int d = e2 & (NDIM - 1);
        int ri = ci[b * NC + i];
        out[e] = h[(b * NPTS + ri) * NDIM + d];
    }
}

extern "C" void kernel_launch(void* const* d_in, const int* in_sizes, int n_in,
                              void* d_out, int out_size, void* d_ws, size_t ws_size,
                              hipStream_t stream) {
    const float* g  = (const float*)d_in[0];   // [4, 2048, 2048]
    const float* h  = (const float*)d_in[1];   // [4, 2048, 256]
    const float* W  = (const float*)d_in[2];   // [256, 1]
    const float* bb = (const float*)d_in[3];   // [1]
    float* out = (float*)d_out;                // new_g (4*1024*1024) ++ new_h (4*1024*256)

    float* scores = (float*)d_ws;                                    // 4*2048 f32
    int*   ci     = (int*)((char*)d_ws + NB * NPTS * sizeof(float)); // 4*1024 i32

    // 512 tiles of 16 rows, one wave each
    scores_wmma_kernel<<<512, 32, 0, stream>>>(h, W, bb, scores);
    // one workgroup per batch
    cluster_kernel<<<NB, 1024, 0, stream>>>(scores, ci);
    // 5,242,880 output elements / 256
    gather_kernel<<<20480, 256, 0, stream>>>(g, h, ci, out);
}
